// CortexBlock_59940563583556
// MI455X (gfx1250) — matrix operand order repaired and checked by reading, not compiled, not run on hardware
//
#include <hip/hip_runtime.h>
#include <math.h>
#include <stdint.h>

typedef __attribute__((ext_vector_type(2)))  float  v2f;
typedef __attribute__((ext_vector_type(8)))  float  v8f;
typedef __attribute__((ext_vector_type(16))) __bf16 v16bf;

#define D_MODEL   1024
#define N_HEADS   16
#define D_HEAD    64
#define DECAY     0.95f
#define ALPHA_MAX 0.05f
#define BETA      0.01f

// bf16x3 GEMM tiling
#define BM   64     // block rows
#define BN   128    // block cols
#define KC   32     // K chunk == one bf16 WMMA K
#define APAD 8      // LDS row pad (bf16 elems) -> row stride 40 (80B, 16B-aligned)

struct alignas(8) bf16x4 { __bf16 e[4]; };
union  frag16 { float4 f[2]; v16bf v; };

// ---------------------------------------------------------------------------
// fp32 -> (hi, lo) bf16 split: x = hi + lo with hi = rne_bf16(x),
// lo = rne_bf16(x - hi). Vectorized x4.
// ---------------------------------------------------------------------------
__global__ __launch_bounds__(256)
void cvt_bf16x2_kernel(const float* __restrict__ x, __bf16* __restrict__ hi,
                       __bf16* __restrict__ lo, int n4)
{
    const int i = blockIdx.x * 256 + threadIdx.x;
    if (i >= n4) return;
    const float4 a = ((const float4*)x)[i];
    bf16x4 ph, pl;
    ph.e[0] = (__bf16)a.x; pl.e[0] = (__bf16)(a.x - (float)ph.e[0]);
    ph.e[1] = (__bf16)a.y; pl.e[1] = (__bf16)(a.y - (float)ph.e[1]);
    ph.e[2] = (__bf16)a.z; pl.e[2] = (__bf16)(a.z - (float)ph.e[2]);
    ph.e[3] = (__bf16)a.w; pl.e[3] = (__bf16)(a.w - (float)ph.e[3]);
    ((bf16x4*)hi)[i] = ph;
    ((bf16x4*)lo)[i] = pl;
}

// ---------------------------------------------------------------------------
// C[M,N] = A[M,K] @ W[N,K]^T in fp32-accurate bf16x3:
//   C += Ahi*Whi + Ahi*Wlo + Alo*Whi   via V_WMMA_F32_16X16X32_BF16.
// Block = 256 threads = 8 waves, block tile 64x128; wave (wm 0..3, wn 0..1)
// owns a 16x64 strip = 4 accumulators.
// A chunk staged into double-buffered LDS with GLOBAL_LOAD_ASYNC_TO_LDS_B128
// (ASYNCcnt-tracked DMA, overlapped one chunk ahead of the WMMA pipeline).
// WMMAs issued pass-major across the 4 accumulators so dependent WMMAs on the
// same accumulator are 4 apart, filling the CDNA5 WMMA->WMMA hazard window.
// ---------------------------------------------------------------------------
__global__ __launch_bounds__(256)
void gemm_bf16x3_kernel(const __bf16* __restrict__ Ahi, const __bf16* __restrict__ Alo,
                        const __bf16* __restrict__ Whi, const __bf16* __restrict__ Wlo,
                        float* __restrict__ C, int M, int N, int K)
{
    __shared__ __bf16 AsHi[2][BM][KC + APAD];
    __shared__ __bf16 AsLo[2][BM][KC + APAD];

    const int tid  = threadIdx.x;
    const int wave = tid >> 5;
    const int lane = tid & 31;
    const int row0 = blockIdx.x * BM;
    const int col0 = blockIdx.y * BN;
    const int wm   = wave >> 1;     // 0..3  (M subtile)
    const int wn   = wave & 1;      // 0..1  (N group of 64)

    // 16-bit WMMA fragment addressing (ISA 7.12.2):
    // A: lane -> m = lane&15; k ranges [akb, akb+8) and [16+akb, 16+akb+8), akb = 8*(lane>=16)
    // B: lane -> n = lane&15; k ranges [bkb, bkb+8) and [16+bkb, 16+bkb+8), bkb = 8*(lane>=16)
    const int fm  = lane & 15;
    const int akb = (lane >> 4) << 3;
    const int bkb = (lane >> 4) << 3;

    v8f acc[4];
    #pragma unroll
    for (int nt = 0; nt < 4; ++nt)
        acc[nt] = (v8f){0.f,0.f,0.f,0.f,0.f,0.f,0.f,0.f};

    // cooperative A staging: 64x32 bf16 per part, 8 bf16 (16B) per thread per part
    const int sr = tid >> 2;            // 0..63
    const int sc = (tid & 3) << 3;      // 0,8,16,24
    const __bf16* ahg = Ahi + (size_t)(row0 + sr) * K + sc;
    const __bf16* alg = Alo + (size_t)(row0 + sr) * K + sc;

    // per-lane LDS byte offsets (flat LDS aperture: addr[31:0] == LDS offset)
    const uint32_t dsthi[2] = { (uint32_t)(uintptr_t)&AsHi[0][sr][sc],
                                (uint32_t)(uintptr_t)&AsHi[1][sr][sc] };
    const uint32_t dstlo[2] = { (uint32_t)(uintptr_t)&AsLo[0][sr][sc],
                                (uint32_t)(uintptr_t)&AsLo[1][sr][sc] };

    const int am = (wm << 4) + fm;      // LDS row for this lane's A fragment

    const __bf16* wh[4];
    const __bf16* wl[4];
    #pragma unroll
    for (int nt = 0; nt < 4; ++nt) {
        const int col = col0 + (wn << 6) + (nt << 4) + fm;
        wh[nt] = Whi + (size_t)col * K + bkb;
        wl[nt] = Wlo + (size_t)col * K + bkb;
    }

    // async DMA: 16B per lane per part into the selected LDS buffer
    auto issue_stage = [&](int b, int kc) {
        asm volatile("global_load_async_to_lds_b128 %0, %1, off"
                     :: "v"(dsthi[b]), "v"(ahg + kc) : "memory");
        asm volatile("global_load_async_to_lds_b128 %0, %1, off"
                     :: "v"(dstlo[b]), "v"(alg + kc) : "memory");
    };

    issue_stage(0, 0);                   // prologue: chunk 0 -> buffer 0
    int buf = 0;
    for (int kc = 0; kc < K; kc += KC, buf ^= 1) {
        const bool more = (kc + KC) < K;
        if (more) {
            issue_stage(buf ^ 1, kc + KC);           // prefetch next chunk
            // async loads complete in order: <=2 outstanding => current landed
            asm volatile("s_wait_asynccnt 2" ::: "memory");
        } else {
            asm volatile("s_wait_asynccnt 0" ::: "memory");
        }
        __syncthreads();

        frag16 ahif, alof;
        ahif.f[0] = *(const float4*)&AsHi[buf][am][akb];
        ahif.f[1] = *(const float4*)&AsHi[buf][am][16 + akb];
        alof.f[0] = *(const float4*)&AsLo[buf][am][akb];
        alof.f[1] = *(const float4*)&AsLo[buf][am][16 + akb];

        // load all 8 B fragments for this chunk (clause-friendly)
        frag16 bh[4], bl[4];
        #pragma unroll
        for (int nt = 0; nt < 4; ++nt) {
            bh[nt].f[0] = *(const float4*)(wh[nt] + kc);
            bh[nt].f[1] = *(const float4*)(wh[nt] + kc + 16);
            bl[nt].f[0] = *(const float4*)(wl[nt] + kc);
            bl[nt].f[1] = *(const float4*)(wl[nt] + kc + 16);
        }

        // pass-major issue: dependent WMMAs on one accumulator are 4 apart
        #pragma unroll
        for (int nt = 0; nt < 4; ++nt)
            acc[nt] = __builtin_amdgcn_wmma_f32_16x16x32_bf16(
                false, ahif.v, false, bh[nt].v, (short)0, acc[nt], false, false);
        #pragma unroll
        for (int nt = 0; nt < 4; ++nt)
            acc[nt] = __builtin_amdgcn_wmma_f32_16x16x32_bf16(
                false, ahif.v, false, bl[nt].v, (short)0, acc[nt], false, false);
        #pragma unroll
        for (int nt = 0; nt < 4; ++nt)
            acc[nt] = __builtin_amdgcn_wmma_f32_16x16x32_bf16(
                false, alof.v, false, bh[nt].v, (short)0, acc[nt], false, false);

        __syncthreads();   // readers done before next iteration's prefetch overwrites buf
    }

    // D layout: VGPR j -> m = j + 8*(lane>=16), n = lane&15
    const int cm = (lane >> 4) << 3;
    const int cn = lane & 15;
    #pragma unroll
    for (int nt = 0; nt < 4; ++nt) {
        float* cp = C + (size_t)(row0 + (wm << 4) + cm) * N
                      + (col0 + (wn << 6) + (nt << 4) + cn);
        #pragma unroll
        for (int j = 0; j < 8; ++j) cp[(size_t)j * N] = acc[nt][j];
    }
}

// ---------------------------------------------------------------------------
// alpha = min(sigmoid(x @ Wa^T + ba) * m_gate * alpha_scale, ALPHA_MAX)
// N=16 == one WMMA tile; fp32 WMMA keeps full precision on this tiny GEMM.
// ---------------------------------------------------------------------------
__global__ __launch_bounds__(32)
void alpha_proj_kernel(const float* __restrict__ X, const float* __restrict__ Wa,
                       const float* __restrict__ ba, const float* __restrict__ m_gate,
                       const float* __restrict__ alpha_scale,
                       float* __restrict__ alpha_out, int M, int K)
{
    const int lane = threadIdx.x;
    const int row0 = blockIdx.x << 4;
    const int fm   = lane & 15;
    const int fkb  = (lane >> 4) << 1;

    v8f acc = {0.f, 0.f, 0.f, 0.f, 0.f, 0.f, 0.f, 0.f};
    const float* arow = X  + (size_t)(row0 + fm) * K;
    const float* wrow = Wa + (size_t)fm * K;

    for (int k = 0; k < K; k += 4) {
        v2f af, bf;
        const float2 av = *(const float2*)(arow + k + fkb);
        const float2 wv = *(const float2*)(wrow + k + fkb);
        af.x = av.x; af.y = av.y;
        bf.x = wv.x; bf.y = wv.y;
        acc = __builtin_amdgcn_wmma_f32_16x16x4_f32(
            false, af, false, bf, (short)0, acc, false, false);
    }

    const int cm = (lane >> 4) << 3;
    const int h  = lane & 15;
    const float bah = ba[h];
    #pragma unroll
    for (int j = 0; j < 8; ++j) {
        const int row = row0 + cm + j;
        const float logit = acc[j] + bah;
        const float sg = 1.f / (1.f + __expf(-logit));
        const float av = sg * m_gate[row] * alpha_scale[(size_t)row * N_HEADS + h];
        alpha_out[(size_t)row * N_HEADS + h] = fminf(av, ALPHA_MAX);
    }
}

// ---------------------------------------------------------------------------
// Sequential fast-weight scan. One 64-thread block per (b,h) chain.
// Thread (g = tid>>4, r = tid&15) keeps U[d][r] and V[r][d] for
// d in [16g, 16g+16) entirely in registers (32 floats/thread).
// ---------------------------------------------------------------------------
__global__ __launch_bounds__(64)
void fastweight_scan_kernel(const float* __restrict__ q, const float* __restrict__ k,
                            const float* __restrict__ v, const float* __restrict__ alpha,
                            const float* __restrict__ mix_logit,
                            float* __restrict__ y, int B, int T)
{
    const int bh  = blockIdx.x;     // 0..63
    const int b   = bh >> 4;
    const int h   = bh & 15;
    const int tid = threadIdx.x;    // 0..63
    const int r   = tid & 15;
    const int g   = tid >> 4;
    const int gb  = g << 4;

    __shared__ float sk[64], sv[64], sq[64];
    __shared__ float red4[4][16];
    __shared__ float reds[8];
    __shared__ float sa;

    float U[16], Vt[16];
    #pragma unroll
    for (int i = 0; i < 16; ++i) { U[i] = 0.f; Vt[i] = 0.f; }

    const float ml = mix_logit[h];
    const float inv_sqrt = 0.125f;  // 1/sqrt(64)
    const size_t rowBase = (size_t)b * T;
    const size_t colOff  = (size_t)h * D_HEAD;

    for (int t = 0; t < T; ++t) {
        const size_t off = (rowBase + t) * D_MODEL + colOff;
        sk[tid] = k[off + tid];
        sv[tid] = v[off + tid];
        sq[tid] = q[off + tid];
        if (tid == 0) sa = alpha[(rowBase + t) * N_HEADS + h];
        __syncthreads();

        const float a = sa;
        float p = 0.f;
        #pragma unroll
        for (int i = 0; i < 16; ++i) {
            U[i]  *= DECAY;
            Vt[i] *= DECAY;
            p += sk[gb + i] * U[i];
        }
        red4[g][r] = p;
        __syncthreads();
        const float ku  = red4[0][r] + red4[1][r] + red4[2][r] + red4[3][r];
        const float aku = a * ku;

        float kf[16];
        #pragma unroll
        for (int i = 0; i < 16; ++i) {
            U[i]  += aku * sk[gb + i];
            Vt[i] += aku * sv[gb + i];
            U[i]  -= BETA * fminf(fmaxf(U[i],  -1.f), 1.f);
            Vt[i] -= BETA * fminf(fmaxf(Vt[i], -1.f), 1.f);
            kf[i] = U[i] * Vt[i];
        }
        #pragma unroll
        for (int msk = 1; msk < 16; msk <<= 1) {
            #pragma unroll
            for (int i = 0; i < 16; ++i)
                kf[i] += __shfl_xor(kf[i], msk, 32);
        }

        float ps = 0.f, pf = 0.f;
        #pragma unroll
        for (int i = 0; i < 16; ++i) {
            const float qn = sq[gb + i] * inv_sqrt;
            ps += qn * sk[gb + i];
            pf += qn * kf[i];
        }
        __syncthreads();
        if (r == 0) { reds[g] = ps; reds[4 + g] = pf; }
        __syncthreads();
        const float ss = reds[0] + reds[1] + reds[2] + reds[3];
        const float sf = reds[4] + reds[5] + reds[6] + reds[7];

        const float l0 = ss + ml, l1 = sf + ml;
        const float mx = fmaxf(l0, l1);
        const float e0 = __expf(l0 - mx), e1 = __expf(l1 - mx);
        const float inv = 1.f / (e0 + e1);

        y[off + gb + r] = (e0 * inv) * sv[gb + r] + (e1 * inv) * kf[r];
        __syncthreads();
    }
}

// ---------------------------------------------------------------------------
extern "C" void kernel_launch(void* const* d_in, const int* in_sizes, int n_in,
                              void* d_out, int out_size, void* d_ws, size_t ws_size,
                              hipStream_t stream)
{
    (void)in_sizes; (void)n_in; (void)out_size; (void)ws_size;
    const float* hs          = (const float*)d_in[0];
    const float* m_gate      = (const float*)d_in[1];
    const float* alpha_scale = (const float*)d_in[2];
    const float* Wq          = (const float*)d_in[3];
    const float* Wk          = (const float*)d_in[4];
    const float* Wv          = (const float*)d_in[5];
    const float* Wo          = (const float*)d_in[6];
    const float* Wa          = (const float*)d_in[7];
    const float* ba          = (const float*)d_in[8];
    const float* mix_logit   = (const float*)d_in[9];

    const int B = 4, T = 2048;
    const int M = B * T;                        // 8192
    const size_t MD = (size_t)M * D_MODEL;      // 8388608
    const size_t DD = (size_t)D_MODEL * D_MODEL;

    // workspace layout: fp32 {q,k,v,alpha,y} then bf16 {xhi,xlo,whi[4],wlo[4]}
    float* q  = (float*)d_ws;
    float* k  = q  + MD;
    float* v  = k  + MD;
    float* al = v  + MD;
    float* y  = al + (size_t)M * N_HEADS;
    __bf16* xhi = (__bf16*)(y + MD);            // reused for y-split later
    __bf16* xlo = xhi + MD;
    __bf16* whi = xlo + MD;                     // 4 weight matrices, packed
    __bf16* wlo = whi + 4 * DD;

    const dim3 cblk(256);
    // split hidden_states and the 4 big weight matrices into bf16 hi/lo
    cvt_bf16x2_kernel<<<dim3((unsigned)(MD / 4 / 256)), cblk, 0, stream>>>(hs, xhi, xlo, (int)(MD / 4));
    cvt_bf16x2_kernel<<<dim3((unsigned)(DD / 4 / 256)), cblk, 0, stream>>>(Wq, whi + 0 * DD, wlo + 0 * DD, (int)(DD / 4));
    cvt_bf16x2_kernel<<<dim3((unsigned)(DD / 4 / 256)), cblk, 0, stream>>>(Wk, whi + 1 * DD, wlo + 1 * DD, (int)(DD / 4));
    cvt_bf16x2_kernel<<<dim3((unsigned)(DD / 4 / 256)), cblk, 0, stream>>>(Wv, whi + 2 * DD, wlo + 2 * DD, (int)(DD / 4));
    cvt_bf16x2_kernel<<<dim3((unsigned)(DD / 4 / 256)), cblk, 0, stream>>>(Wo, whi + 3 * DD, wlo + 3 * DD, (int)(DD / 4));

    const dim3 gblk(256);
    const dim3 grd(M / BM, D_MODEL / BN);       // (128, 8)
    gemm_bf16x3_kernel<<<grd, gblk, 0, stream>>>(xhi, xlo, whi + 0 * DD, wlo + 0 * DD, q, M, D_MODEL, D_MODEL);
    gemm_bf16x3_kernel<<<grd, gblk, 0, stream>>>(xhi, xlo, whi + 1 * DD, wlo + 1 * DD, k, M, D_MODEL, D_MODEL);
    gemm_bf16x3_kernel<<<grd, gblk, 0, stream>>>(xhi, xlo, whi + 2 * DD, wlo + 2 * DD, v, M, D_MODEL, D_MODEL);

    alpha_proj_kernel<<<dim3(M / 16), dim3(32), 0, stream>>>(
        hs, Wa, ba, m_gate, alpha_scale, al, M, D_MODEL);

    fastweight_scan_kernel<<<dim3(B * N_HEADS), dim3(64), 0, stream>>>(
        q, k, v, al, mix_logit, y, B, T);

    // split y (reusing the x split buffers) and run the output projection
    cvt_bf16x2_kernel<<<dim3((unsigned)(MD / 4 / 256)), cblk, 0, stream>>>(y, xhi, xlo, (int)(MD / 4));
    gemm_bf16x3_kernel<<<grd, gblk, 0, stream>>>(xhi, xlo, whi + 3 * DD, wlo + 3 * DD,
                                                 (float*)d_out, M, D_MODEL, D_MODEL);
}
